// Mamba2Mixer_79173427134759
// MI455X (gfx1250) — matrix operand order repaired and checked by reading, not compiled, not run on hardware
//
#include <hip/hip_runtime.h>
#include <hip/hip_bf16.h>
#include <cstdint>

// ---------------------------------------------------------------- constants
constexpr int D_MODEL   = 2048;
constexpr int D_STATE   = 128;
constexpr int D_CONV    = 4;
constexpr int NHEADS    = 64;
constexpr int NGROUPS   = 8;
constexpr int HEAD_DIM  = 64;
constexpr int CHUNK     = 256;
constexpr int SEQLEN    = 2048;
constexpr float EPS     = 1e-5f;
constexpr int D_INNER   = NHEADS * HEAD_DIM;                    // 4096
constexpr int CONV_DIM  = D_INNER + 2 * NGROUPS * D_STATE;      // 6144
constexpr int D_IN_PROJ = 2 * D_INNER + 2 * NGROUPS * D_STATE + NHEADS; // 10304
constexpr int NCHUNK    = SEQLEN / CHUNK;                       // 8
constexpr int GSIZE     = D_INNER / NGROUPS;                    // 512

#define DEV __device__ __forceinline__

typedef __attribute__((ext_vector_type(16))) __bf16 v16bf;
typedef __attribute__((ext_vector_type(8)))  float  v8f;
typedef __attribute__((ext_vector_type(2)))  float  v2f;

DEV v8f zero8() { v8f z = {0.f,0.f,0.f,0.f,0.f,0.f,0.f,0.f}; return z; }

DEV v8f wmma_bf16(v16bf a, v16bf b, v8f c) {
  return __builtin_amdgcn_wmma_f32_16x16x32_bf16(false, a, false, b, (short)0, c, false, false);
}
DEV v8f wmma_f32(v2f a, v2f b, v8f c) {
  return __builtin_amdgcn_wmma_f32_16x16x4_f32(false, a, false, b, (short)0, c, false, false);
}

DEV float silu(float x)     { return x / (1.f + __expf(-x)); }
DEV float softplus(float x) { return (x > 20.f) ? x : log1pf(__expf(x)); }

// ---------------------------------------------------------------- f32 -> bf16
__global__ void cvt_bf16_kernel(const float* __restrict__ src, __bf16* __restrict__ dst, int n) {
  int stride = gridDim.x * blockDim.x;
  for (int i = blockIdx.x * blockDim.x + threadIdx.x; i < n; i += stride)
    dst[i] = (__bf16)src[i];
}

// ---------------------------------------------------------------- bf16 GEMM: C[M,N] = A[M,K] * B[N,K]^T
// Block (256 thr = 8 waves) computes 128(M) x 64(N). The shared B panel
// (64 cols x 32 K = 4KB bf16) is staged into LDS with async loads
// (global_load_async_to_lds_b128), double-buffered across K-steps, so B is
// fetched from global once per block instead of once per wave (8x cut).
// A fragments are per-wave global b128 loads, register-prefetched one
// K-step ahead. All four B fragments are ds-loaded before the WMMA chain so
// the four v_wmma issue back-to-back behind a single dscnt wait.
__global__ __launch_bounds__(256) void gemm_bf16_nt(const __bf16* __restrict__ A,
                                                    const __bf16* __restrict__ B,
                                                    float* __restrict__ C,
                                                    int M, int N, int K) {
  __shared__ __align__(16) unsigned char Bsh[2][64 * 64];  // [buf][n(64) x 64B (32 bf16)]
  const int tid  = threadIdx.x;
  const int wid  = tid >> 5;
  const int lane = tid & 31;
  const int nnb  = N >> 6;
  const int mb   = blockIdx.x / nnb;
  const int nb   = blockIdx.x - mb * nnb;
  const int m0   = (mb << 7) + (wid << 4);
  const int n0   = nb << 6;

  // async staging: thread t fills 16B chunk (t&3) of B row n0+(t>>2)
  const int srow = tid >> 2, schunk = tid & 3;
  const __bf16* gB = B + (size_t)(n0 + srow) * K + (schunk << 3);
  const unsigned int lds0 =
      (unsigned int)(uintptr_t)(&Bsh[0][(srow << 6) + (schunk << 4)]);
  const unsigned int lds1 =
      (unsigned int)(uintptr_t)(&Bsh[1][(srow << 6) + (schunk << 4)]);

  // prologue: stage K-step 0 into buffer 0
  asm volatile("global_load_async_to_lds_b128 %0, %1, off"
               :: "v"(lds0), "v"(gB) : "memory");

  const int kA = (lane >> 4) << 3;  // A frag: halves kA..kA+7 and kA+16..kA+23
  const __bf16* arow = A + (size_t)(m0 + (lane & 15)) * K;

  union AU { uint4 u[2]; v16bf v; };
  AU ua;  // current A fragment (prefetched)
  {
    const __bf16* pa = arow + kA;
    ua.u[0] = *(const uint4*)(pa);
    ua.u[1] = *(const uint4*)(pa + 16);
  }

  v8f acc[4] = { zero8(), zero8(), zero8(), zero8() };

  for (int k0 = 0; k0 < K; k0 += 32) {
    const int cur = (k0 >> 5) & 1;
    // my async stores for buffer `cur` have landed:
    asm volatile("s_wait_asynccnt 0x0" ::: "memory");
    // everyone's have landed; previous compute (reader of buf cur^1) done:
    __syncthreads();
    if (k0 + 32 < K) {  // stage next K-step into the other buffer
      const unsigned int dst = cur ? lds0 : lds1;
      const __bf16* gnext = gB + k0 + 32;
      asm volatile("global_load_async_to_lds_b128 %0, %1, off"
                   :: "v"(dst), "v"(gnext) : "memory");
    }
    // prefetch next A fragment while WMMAs run
    AU un;
    if (k0 + 32 < K) {
      const __bf16* pa = arow + k0 + 32 + kA;
      un.u[0] = *(const uint4*)(pa);
      un.u[1] = *(const uint4*)(pa + 16);
    }
    // B fragments from LDS: row n = 16j + (lane&15), k = 8*(lane>>4)..+7 and 16+8*(lane>>4)..+7
    const unsigned char* bb = &Bsh[cur][((lane & 15) << 6) + ((lane >> 4) << 4)];
    union { uint4 u[2]; v16bf v; } ub[4];
#pragma unroll
    for (int j = 0; j < 4; ++j) {
      const unsigned char* bp = bb + (j << 10);  // 16 rows * 64B per n-tile
      ub[j].u[0] = *(const uint4*)(bp);
      ub[j].u[1] = *(const uint4*)(bp + 32);
    }
#pragma unroll
    for (int j = 0; j < 4; ++j)
      acc[j] = wmma_bf16(ua.v, ub[j].v, acc[j]);
    ua = un;
  }

  const int crow = m0 + ((lane >> 4) << 3);
#pragma unroll
  for (int j = 0; j < 4; ++j) {
    float* cp = C + (size_t)crow * N + n0 + 16 * j + (lane & 15);
#pragma unroll
    for (int r = 0; r < 8; ++r) cp[(size_t)r * N] = acc[j][r];
  }
}

// ---------------------------------------------------------------- depthwise conv4 + SiLU
__global__ void conv_silu_kernel(const float* __restrict__ zx, const float* __restrict__ cw,
                                 const float* __restrict__ cb, float* __restrict__ xbc) {
  int idx = blockIdx.x * blockDim.x + threadIdx.x;
  if (idx >= SEQLEN * CONV_DIM) return;
  int t = idx / CONV_DIM, c = idx - t * CONV_DIM;
  float s = cb[c];
#pragma unroll
  for (int k = 0; k < D_CONV; ++k) {
    int tt = t + k - (D_CONV - 1);
    if (tt >= 0) s += zx[(size_t)tt * D_IN_PROJ + D_INNER + c] * cw[c * D_CONV + k];
  }
  xbc[idx] = silu(s);
}

// ---------------------------------------------------------------- dt = softplus(raw + bias)
__global__ void dt_kernel(const float* __restrict__ zx, const float* __restrict__ dt_bias,
                          float* __restrict__ dt) {
  int idx = blockIdx.x * blockDim.x + threadIdx.x;
  if (idx >= SEQLEN * NHEADS) return;
  int t = idx >> 6, h = idx & 63;
  dt[idx] = softplus(zx[(size_t)t * D_IN_PROJ + D_INNER + CONV_DIM + h] + dt_bias[h]);
}

// ---------------------------------------------------------------- per-(chunk,head) cumsum of dt*A
__global__ void cumsum_kernel(const float* __restrict__ dt, const float* __restrict__ Ap,
                              float* __restrict__ dAcs) {
  int idx = blockIdx.x * blockDim.x + threadIdx.x;
  if (idx >= NCHUNK * NHEADS) return;
  int c = idx >> 6, h = idx & 63;
  float a = Ap[h], run = 0.f;
  for (int s = 0; s < CHUNK; ++s) {
    size_t th = (size_t)(c * CHUNK + s) * NHEADS + h;
    run += dt[th] * a;
    dAcs[th] = run;
  }
}

// ---------------------------------------------------------------- CB[c,g,s,t] = sum_n C[s,n]*B[t,n]  (f32 WMMA)
__global__ __launch_bounds__(256) void cb_kernel(const float* __restrict__ xbc,
                                                 float* __restrict__ CB) {
  const int wave = blockIdx.x * 8 + (threadIdx.x >> 5);
  const int lane = threadIdx.x & 31;
  const int cg = wave >> 8, tile = wave & 255;
  const int c = cg >> 3, g = cg & 7;
  const int s0 = (tile >> 4) << 4, t0 = (tile & 15) << 4;
  const int klane = (lane >> 4) << 1;
  const float* Crow = xbc + (size_t)(c * CHUNK + s0 + (lane & 15)) * CONV_DIM
                      + (D_INNER + NGROUPS * D_STATE + g * D_STATE);
  const float* Brow = xbc + (size_t)(c * CHUNK + t0 + (lane & 15)) * CONV_DIM
                      + (D_INNER + g * D_STATE);
  v8f acc = zero8();
  for (int k0 = 0; k0 < D_STATE; k0 += 4) {
    float2 qa = *(const float2*)(Crow + k0 + klane);
    float2 qb = *(const float2*)(Brow + k0 + klane);
    v2f a = { qa.x, qa.y }, b = { qb.x, qb.y };
    acc = wmma_f32(a, b, acc);
  }
  const int srow = s0 + ((lane >> 4) << 3);
  float* outp = CB + ((size_t)cg * CHUNK + srow) * CHUNK + t0 + (lane & 15);
#pragma unroll
  for (int r = 0; r < 8; ++r) outp[(size_t)r * CHUNK] = acc[r];
}

// ---------------------------------------------------------------- y[s,h,p] = sum_{t<=s} CB*exp(dAcs_s-dAcs_t)*dt_t * x[t,p]
__global__ __launch_bounds__(256) void ydiag_kernel(const float* __restrict__ xbc,
                                                    const float* __restrict__ CB,
                                                    const float* __restrict__ dAcs,
                                                    const float* __restrict__ dt,
                                                    float* __restrict__ Y) {
  const int wave = blockIdx.x * 8 + (threadIdx.x >> 5);
  const int lane = threadIdx.x & 31;
  const int pt = wave & 3, st = (wave >> 2) & 15, ch = wave >> 6;
  const int c = ch >> 6, h = ch & 63, g = h >> 3;
  const int s0 = st << 4, p0 = pt << 4;
  const int sA = s0 + (lane & 15);
  const int pB = p0 + (lane & 15);
  const int klane = (lane >> 4) << 1;
  const float dacs_s = dAcs[(size_t)(c * CHUNK + sA) * NHEADS + h];
  const float* cbrow = CB + ((size_t)(c * NGROUPS + g) * CHUNK + sA) * CHUNK;
  v8f acc = zero8();
  for (int t0 = 0; t0 < s0 + 16; t0 += 4) {
    const int ta = t0 + klane;
    v2f a;
#pragma unroll
    for (int j = 0; j < 2; ++j) {
      const int t = ta + j;
      const size_t th = (size_t)(c * CHUNK + t) * NHEADS + h;
      float v = cbrow[t] * __expf(dacs_s - dAcs[th]) * dt[th];
      a[j] = (t <= sA) ? v : 0.f;   // causal mask (select, in-bounds loads)
    }
    const float* xr = xbc + (size_t)(c * CHUNK + ta) * CONV_DIM + h * HEAD_DIM + pB;
    v2f b = { xr[0], xr[CONV_DIM] };
    acc = wmma_f32(a, b, acc);
  }
  const int srow = s0 + ((lane >> 4) << 3);
  float* yp = Y + (size_t)(c * CHUNK + srow) * D_INNER + h * HEAD_DIM + pB;
#pragma unroll
  for (int r = 0; r < 8; ++r) yp[(size_t)r * D_INNER] = acc[r];
}

// ---------------------------------------------------------------- states[c,h,p,n] = sum_t w_t*x[t,p]*B[t,n]
__global__ __launch_bounds__(256) void states_kernel(const float* __restrict__ xbc,
                                                     const float* __restrict__ dAcs,
                                                     const float* __restrict__ dt,
                                                     float* __restrict__ S) {
  const int wave = blockIdx.x * 8 + (threadIdx.x >> 5);
  const int lane = threadIdx.x & 31;
  const int ntl = wave & 7, pt = (wave >> 3) & 3, ch = wave >> 5;
  const int c = ch >> 6, h = ch & 63, g = h >> 3;
  const int p0 = pt << 4, n0 = ntl << 4;
  const int pA = p0 + (lane & 15);
  const int nB = n0 + (lane & 15);
  const int klane = (lane >> 4) << 1;
  const float dacs_last = dAcs[(size_t)(c * CHUNK + CHUNK - 1) * NHEADS + h];
  v8f acc = zero8();
  for (int t0 = 0; t0 < CHUNK; t0 += 4) {
    const int ta = t0 + klane;
    v2f a;
#pragma unroll
    for (int j = 0; j < 2; ++j) {
      const int t = ta + j;
      const size_t th = (size_t)(c * CHUNK + t) * NHEADS + h;
      a[j] = __expf(dacs_last - dAcs[th]) * dt[th]
             * xbc[(size_t)(c * CHUNK + t) * CONV_DIM + h * HEAD_DIM + pA];
    }
    const float* br = xbc + (size_t)(c * CHUNK + ta) * CONV_DIM + D_INNER + g * D_STATE + nB;
    v2f b = { br[0], br[CONV_DIM] };
    acc = wmma_f32(a, b, acc);
  }
  const int prow = p0 + ((lane >> 4) << 3);
  float* sp = S + ((size_t)(c * NHEADS + h) * HEAD_DIM + prow) * D_STATE + nB;
#pragma unroll
  for (int r = 0; r < 8; ++r) sp[(size_t)r * D_STATE] = acc[r];
}

// ---------------------------------------------------------------- inter-chunk scan (in place: states -> prev)
__global__ void scan_kernel(const float* __restrict__ dAcs, float* __restrict__ S) {
  int idx = blockIdx.x * blockDim.x + threadIdx.x;
  if (idx >= NHEADS * HEAD_DIM * D_STATE) return;
  int h = idx / (HEAD_DIM * D_STATE);
  int rem = idx - h * HEAD_DIM * D_STATE;   // p*128 + n
  float carry = 0.f;
  for (int c = 0; c < NCHUNK; ++c) {
    size_t off = (size_t)(c * NHEADS + h) * HEAD_DIM * D_STATE + rem;
    float st  = S[off];
    float dec = __expf(dAcs[(size_t)(c * CHUNK + CHUNK - 1) * NHEADS + h]);
    S[off] = carry;                 // prev state for chunk c
    carry = carry * dec + st;
  }
}

// ---------------------------------------------------------------- y += exp(dAcs_s) * C[s,:] . prev[h,p,:]
__global__ __launch_bounds__(256) void yfix_kernel(const float* __restrict__ xbc,
                                                   const float* __restrict__ S,
                                                   const float* __restrict__ dAcs,
                                                   float* __restrict__ Y) {
  const int wave = blockIdx.x * 8 + (threadIdx.x >> 5);
  const int lane = threadIdx.x & 31;
  const int pt = wave & 3, st = (wave >> 2) & 15, ch = wave >> 6;
  const int c = ch >> 6, h = ch & 63, g = h >> 3;
  const int s0 = st << 4, p0 = pt << 4;
  const int pB = p0 + (lane & 15);
  const int klane = (lane >> 4) << 1;
  const float* crow = xbc + (size_t)(c * CHUNK + s0 + (lane & 15)) * CONV_DIM
                      + (D_INNER + NGROUPS * D_STATE + g * D_STATE);
  const float* prow = S + ((size_t)(c * NHEADS + h) * HEAD_DIM + pB) * D_STATE;
  v8f acc = zero8();
  for (int k0 = 0; k0 < D_STATE; k0 += 4) {
    float2 qa = *(const float2*)(crow + k0 + klane);
    float2 qb = *(const float2*)(prow + k0 + klane);
    v2f a = { qa.x, qa.y }, b = { qb.x, qb.y };
    acc = wmma_f32(a, b, acc);
  }
  const int srow = s0 + ((lane >> 4) << 3);
  float* yp = Y + (size_t)(c * CHUNK + srow) * D_INNER + h * HEAD_DIM + pB;
#pragma unroll
  for (int r = 0; r < 8; ++r) {
    float f = __expf(dAcs[(size_t)(c * CHUNK + srow + r) * NHEADS + h]);
    yp[(size_t)r * D_INNER] += acc[r] * f;
  }
}

// ---------------------------------------------------------------- D-skip + SiLU gate + group RMS norm -> bf16
__global__ __launch_bounds__(256) void gatenorm_kernel(const float* __restrict__ Y,
                                                       const float* __restrict__ xbc,
                                                       const float* __restrict__ zx,
                                                       const float* __restrict__ Dp,
                                                       const float* __restrict__ nw,
                                                       __bf16* __restrict__ Ybf) {
  const int t = blockIdx.x;
  const int tid = threadIdx.x;
  const int d0 = tid * 16;          // 16 elems / thread; wave(32) covers one group of 512
  const int h = d0 >> 6;
  const float dp = Dp[h];
  float v[16];
  float ss = 0.f;
#pragma unroll
  for (int i = 0; i < 16; ++i) {
    int d = d0 + i;
    float y = Y[(size_t)t * D_INNER + d] + dp * xbc[(size_t)t * CONV_DIM + d];
    float z = zx[(size_t)t * D_IN_PROJ + d];
    y *= silu(z);
    v[i] = y;
    ss += y * y;
  }
#pragma unroll
  for (int o = 16; o > 0; o >>= 1) ss += __shfl_xor(ss, o, 32);
  const float r = rsqrtf(ss * (1.f / GSIZE) + EPS);
#pragma unroll
  for (int i = 0; i < 16; ++i)
    Ybf[(size_t)t * D_INNER + d0 + i] = (__bf16)(v[i] * r * nw[d0 + i]);
}

// ---------------------------------------------------------------- launcher
extern "C" void kernel_launch(void* const* d_in, const int* in_sizes, int n_in,
                              void* d_out, int out_size, void* d_ws, size_t ws_size,
                              hipStream_t stream) {
  (void)in_sizes; (void)n_in; (void)out_size; (void)ws_size;
  const float* hidden  = (const float*)d_in[0];
  const float* Win     = (const float*)d_in[1];
  const float* convw   = (const float*)d_in[2];
  const float* convb   = (const float*)d_in[3];
  const float* Aparam  = (const float*)d_in[4];
  const float* Dparam  = (const float*)d_in[5];
  const float* dt_bias = (const float*)d_in[6];
  const float* norm_w  = (const float*)d_in[7];
  const float* Wout    = (const float*)d_in[8];
  float* out = (float*)d_out;

  char* ws = (char*)d_ws;
  size_t off = 0;
  auto alloc = [&](size_t bytes) -> char* {
    char* p = ws + off;
    off += (bytes + 255) & ~(size_t)255;
    return p;
  };
  __bf16* hid_bf  = (__bf16*)alloc((size_t)SEQLEN * D_MODEL * 2);
  __bf16* win_bf  = (__bf16*)alloc((size_t)D_IN_PROJ * D_MODEL * 2);
  __bf16* wout_bf = (__bf16*)alloc((size_t)D_MODEL * D_INNER * 2);
  float*  zx      = (float*)alloc((size_t)SEQLEN * D_IN_PROJ * 4);
  float*  xbc     = (float*)alloc((size_t)SEQLEN * CONV_DIM * 4);
  float*  dt      = (float*)alloc((size_t)SEQLEN * NHEADS * 4);
  float*  dAcs    = (float*)alloc((size_t)SEQLEN * NHEADS * 4);
  float*  CB      = (float*)alloc((size_t)NCHUNK * NGROUPS * CHUNK * CHUNK * 4);
  float*  Ybuf    = (float*)alloc((size_t)SEQLEN * D_INNER * 4);
  float*  Sbuf    = (float*)alloc((size_t)NCHUNK * NHEADS * HEAD_DIM * D_STATE * 4);
  __bf16* y_bf    = (__bf16*)alloc((size_t)SEQLEN * D_INNER * 2);

  // 1) fp32 -> bf16 conversions
  cvt_bf16_kernel<<<4096, 256, 0, stream>>>(hidden, hid_bf, SEQLEN * D_MODEL);
  cvt_bf16_kernel<<<4096, 256, 0, stream>>>(Win, win_bf, D_IN_PROJ * D_MODEL);
  cvt_bf16_kernel<<<4096, 256, 0, stream>>>(Wout, wout_bf, D_MODEL * D_INNER);

  // 2) in_proj GEMM: zx[T, D_IN_PROJ] = hidden @ Win^T   (block = 128x64 tile)
  {
    int blocks = (SEQLEN / 128) * (D_IN_PROJ / 64);    // 16*161
    gemm_bf16_nt<<<blocks, 256, 0, stream>>>(hid_bf, win_bf, zx,
                                             SEQLEN, D_IN_PROJ, D_MODEL);
  }

  // 3) conv4 + SiLU; 4) dt; 5) cumsum
  conv_silu_kernel<<<(SEQLEN * CONV_DIM + 255) / 256, 256, 0, stream>>>(zx, convw, convb, xbc);
  dt_kernel<<<(SEQLEN * NHEADS + 255) / 256, 256, 0, stream>>>(zx, dt_bias, dt);
  cumsum_kernel<<<(NCHUNK * NHEADS + 255) / 256, 256, 0, stream>>>(dt, Aparam, dAcs);

  // 6) CB; 7) intra-chunk y; 8) chunk states; 9) scan; 10) inter-chunk correction
  cb_kernel<<<(NCHUNK * NGROUPS * 256) / 8, 256, 0, stream>>>(xbc, CB);
  ydiag_kernel<<<(NCHUNK * NHEADS * 64) / 8, 256, 0, stream>>>(xbc, CB, dAcs, dt, Ybuf);
  states_kernel<<<(NCHUNK * NHEADS * 32) / 8, 256, 0, stream>>>(xbc, dAcs, dt, Sbuf);
  scan_kernel<<<(NHEADS * HEAD_DIM * D_STATE + 255) / 256, 256, 0, stream>>>(dAcs, Sbuf);
  yfix_kernel<<<(NCHUNK * NHEADS * 64) / 8, 256, 0, stream>>>(xbc, Sbuf, dAcs, Ybuf);

  // 11) gate + norm -> bf16
  gatenorm_kernel<<<SEQLEN, 256, 0, stream>>>(Ybuf, xbc, zx, Dparam, norm_w, y_bf);

  // 12) out_proj GEMM: out[T, D_MODEL] = y @ Wout^T
  {
    int blocks = (SEQLEN / 128) * (D_MODEL / 64);      // 16*32
    gemm_bf16_nt<<<blocks, 256, 0, stream>>>(y_bf, wout_bf, out,
                                             SEQLEN, D_MODEL, D_INNER);
  }
}
